// SANIrecursiveLayersModel_54649163875044
// MI455X (gfx1250) — matrix operand order repaired and checked by reading, not compiled, not run on hardware
//
#include <hip/hip_runtime.h>
#include <hip/hip_bf16.h>

// ---------------------------------------------------------------------------
// Problem constants (match reference)
// ---------------------------------------------------------------------------
#define B_  4
#define S_  256
#define H_  1024
#define E_  1024
#define M_  (B_ * S_)      // 1024 token rows
#define KT_ 32             // K / 32 (K = 1024 for both GEMMs)

typedef __attribute__((ext_vector_type(16))) __bf16 v16bf;
typedef __attribute__((ext_vector_type(8)))  float  v8f;
typedef __attribute__((ext_vector_type(4)))  unsigned int v4u;
typedef __attribute__((ext_vector_type(8)))  int v8i;
typedef __attribute__((ext_vector_type(4)))  int v4i;

// Address-space-qualified 128-bit vectors for the async-to-LDS builtin
typedef __attribute__((address_space(1))) v4i v4i_as1;   // global
typedef __attribute__((address_space(3))) v4i v4i_as3;   // LDS

// Feature detection for the CDNA5 data-mover paths (fallbacks keep compiling)
#if defined(__has_builtin)
#  if __has_builtin(__builtin_amdgcn_tensor_load_to_lds) && \
      __has_builtin(__builtin_amdgcn_s_wait_tensorcnt)
#    define HAVE_TDM 1
#  endif
#  if __has_builtin(__builtin_amdgcn_global_load_async_to_lds_b128) && \
      __has_builtin(__builtin_amdgcn_s_wait_asynccnt)
#    define HAVE_ASYNC 1
#  endif
#endif

// (1 - skipLayersDominance) evaluated in double like the reference
__device__ __constant__ float kSkip = (float)(1.0 - 0.9);
#define EPS_ 1e-5f

// ---------------------------------------------------------------------------
// Fragment packing helpers: match CDNA5 WMMA VGPR layouts (CDNA5 ISA 7.12.2)
// A (16x32 bf16): lane = m + (k&8 ? 16 : 0); elem i = (k&7) + (k&16 ? 8 : 0)
// B (32x16 bf16): lane = n + (k&8 ? 16 : 0); elem i = (k&7) + (k&16 ? 8 : 0)
// Each lane's 16 bf16 values are stored contiguously (32B) -> b128-pair loads.
// ---------------------------------------------------------------------------
static __device__ __forceinline__ size_t packA_index(int tm, int tk, int m, int k) {
  int lane = m + ((k & 8) ? 16 : 0);
  int i    = (k & 7) + ((k & 16) ? 8 : 0);
  return ((size_t)(tm * KT_ + tk) * 32 + lane) * 16 + i;
}
static __device__ __forceinline__ size_t packB_index(int tn, int tk, int k, int n) {
  int lane = n + ((k & 8) ? 16 : 0);
  int i    = (k & 7) + ((k & 16) ? 8 : 0);
  return ((size_t)(tn * KT_ + tk) * 32 + lane) * 16 + i;
}

// ---------------------------------------------------------------------------
// init: pipeline flags must be re-zeroed every launch (graph replay safe)
// ---------------------------------------------------------------------------
__global__ void init_kernel(int* __restrict__ ready, int* __restrict__ consumed) {
  int i = threadIdx.x;
  if (i < 128) { ready[i] = 0; consumed[i] = 0; }
}

// ---------------------------------------------------------------------------
// Pack a [1024 x 1024] f32 weight matrix (K x N, row major) into bf16
// B-fragment tiles.
// ---------------------------------------------------------------------------
__global__ void __launch_bounds__(256) pack_B_kernel(const float* __restrict__ W,
                                                     __bf16* __restrict__ out) {
  int idx  = blockIdx.x * 256 + threadIdx.x;   // 1024 blocks -> K*N/4 threads
  int base = idx * 4;
  int k    = base >> 10;
  int n0   = base & 1023;
  float4 w4 = *(const float4*)(W + (size_t)k * 1024 + n0);
  float vals[4] = {w4.x, w4.y, w4.z, w4.w};
  int tk = k >> 5, kk = k & 31;
#pragma unroll
  for (int c = 0; c < 4; ++c) {
    int n = n0 + c;
    out[packB_index(n >> 4, tk, kk, n & 15)] = (__bf16)vals[c];
  }
}

// ---------------------------------------------------------------------------
// Embedding gather: x = emb[labels]  (f32 copy kept for the loss) and
// simultaneously pack bf16 A-fragments for GEMM1.
// ---------------------------------------------------------------------------
__global__ void __launch_bounds__(256) gather_pack_kernel(const int* __restrict__ labels,
                                                          const float* __restrict__ emb,
                                                          float* __restrict__ x,
                                                          __bf16* __restrict__ xpack) {
  int r   = blockIdx.x;            // token row 0..1023
  int lbl = labels[r];
  int k0  = threadIdx.x * 4;
  float4 v = *(const float4*)(emb + (size_t)lbl * E_ + k0);
  *(float4*)(x + (size_t)r * E_ + k0) = v;
  int tm = r >> 4, m = r & 15;
  float vals[4] = {v.x, v.y, v.z, v.w};
#pragma unroll
  for (int c = 0; c < 4; ++c) {
    int k = k0 + c;
    xpack[packA_index(tm, k >> 5, m, k & 31)] = (__bf16)vals[c];
  }
}

// ---------------------------------------------------------------------------
// WMMA GEMM:  C[M x N] = relu(A @ B + bias),  M = N = K = 1024, bf16 in / f32 out.
// Grid: (N/128, M/16); block = 256 threads = 8 waves; one 16x16 tile per wave.
//
// All 8 waves share the same M-tile, so the block's entire pre-swizzled A
// K-strip (KT_*16x32 tiles = 32KB contiguous) is DMA'd into LDS ONCE by the
// Tensor Data Mover (tensor_load_to_lds + s_wait_tensorcnt), then re-read by
// every wave as ds_load_b128 pairs.  B streams from L2 with prefetch.
// ---------------------------------------------------------------------------
__global__ void __launch_bounds__(256) wmma_gemm_bias_relu(const __bf16* __restrict__ Apack,
                                                           const __bf16* __restrict__ Bpack,
                                                           const float* __restrict__ bias,
                                                           float* __restrict__ C, int N) {
  __shared__ __align__(32) char ldsAraw[KT_ * 32 * 16 * 2];   // 32 KB A strip
  int wave = threadIdx.x >> 5;
  int lane = threadIdx.x & 31;
  int tm = blockIdx.y;
  int tn = blockIdx.x * 8 + wave;

  const __bf16* Astrip = Apack + (size_t)tm * KT_ * 32 * 16;

#if HAVE_TDM
  if (threadIdx.x < 32) {   // one TDM descriptor per workgroup (EXEC ignored)
    unsigned ldsOff = (unsigned)(uintptr_t)&ldsAraw[0];       // LDS byte offset
    unsigned long long ga = (unsigned long long)(uintptr_t)Astrip;
    v4u g0 = {};
    g0[0] = 1u;                                               // count=1, user mode
    g0[1] = ldsOff;                                           // lds_addr
    g0[2] = (unsigned)ga;                                     // global_addr[31:0]
    g0[3] = (unsigned)((ga >> 32) & 0x1FFFFFFull) | 0x80000000u; // addr[56:32] | type=2
    v8i g1 = {};
    g1[0] = (3 << 16);                  // workgroup_mask=0, data_size=3 (8B)
    g1[1] = (int)((4096u & 0xFFFFu) << 16);    // tensor_dim0[15:0] in bits 63:48
    g1[2] = (int)((4096u >> 16) | (1u << 16)); // tensor_dim0[31:16] | tensor_dim1 lo = 1
    g1[3] = (int)(4096u << 16);         // tensor_dim1 hi = 0 | tile_dim0 = 4096 (8B units)
    g1[4] = 1;                          // tile_dim1 = 1, tile_dim2 = 0
    g1[5] = 4096;                       // tensor_dim0_stride low 32 (8B units)
    g1[6] = 0;
    g1[7] = 0;
    v4i z4 = {};
    v8i z8 = {};
    __builtin_amdgcn_tensor_load_to_lds(g0, g1, z4, z4, z8, 0);   // 6-arg form
    __builtin_amdgcn_s_wait_tensorcnt(0);
  }
#else
  // Fallback: cooperative vector copy of the 32KB strip
  for (int i = threadIdx.x; i < (KT_ * 32 * 16 * 2) / 16; i += 256)
    ((float4*)ldsAraw)[i] = ((const float4*)Astrip)[i];
#endif
  __syncthreads();

  const v16bf* Alds = (const v16bf*)ldsAraw;
  const v16bf* Bp   = (const v16bf*)Bpack;
  size_t bBase = (size_t)tn * KT_ * 32 + lane;

  v8f acc = {};
#pragma unroll 4
  for (int tk = 0; tk < KT_; ++tk) {
    v16bf a = Alds[tk * 32 + lane];                 // ds_load_b128 x2
    v16bf b = Bp[bBase + (size_t)tk * 32];          // global_load_b128 x2
    if (tk + 1 < KT_)                               // global_prefetch_b8 next B tile
      __builtin_prefetch(&Bp[bBase + (size_t)(tk + 1) * 32], 0, 1);
    acc = __builtin_amdgcn_wmma_f32_16x16x32_bf16(false, a, false, b,
                                                  (short)0, acc, false, false);
  }

  // C/D layout: VGPR v -> row m = v + 8*(lane>=16), col n = lane & 15
  int n    = tn * 16 + (lane & 15);
  int mrow = tm * 16 + ((lane >> 4) << 3);
  float bv = bias[n];
#pragma unroll
  for (int v = 0; v < 8; ++v) {
    float val = acc[v] + bv;
    C[(size_t)(mrow + v) * N + n] = fmaxf(val, 0.0f);
  }
}

// ---------------------------------------------------------------------------
// 256-step recurrent scan, pipelined across workgroups.
//  - 128 WGs x 8 waves; wave w owns row r = 8*wg + w (t = r % 256).
//  - Rows double-buffered in LDS (2 x 8 x 1024 f32) + 1 staging row = 68KB.
//  - Jacobi step per reference:
//      cat = [prev | curr];  out[h] = relu(W[h,0]*cat[2h] + W[h,1]*cat[2h+1] + b[h])
//      val = out*(1-DOM) + hidden;  hidden' = LayerNorm(val)
//    (h < 512 reads the *previous* row; h >= 512 reads the *own* row)
//  - WG boundary row flows through an L2-resident 2-deep ring with
//    acquire/release flags -> no grid-wide barriers, pure pipeline.
//  - Neighbor staging uses global_load_async_to_lds_b128 (ASYNCcnt) when
//    available: data never touches VGPRs.
// ---------------------------------------------------------------------------
__global__ void __launch_bounds__(256) scan_kernel(const float* __restrict__ h0,
                                                   const float* __restrict__ Wsani,  // [H,2]
                                                   const float* __restrict__ bsani,
                                                   const float* __restrict__ lnw,
                                                   const float* __restrict__ lnb,
                                                   float* __restrict__ ring,         // [2][128][H]
                                                   int* __restrict__ ready,
                                                   int* __restrict__ consumed,
                                                   __bf16* __restrict__ hpack) {
  extern __shared__ float lds[];
  float* buf0    = lds;                 // [8][1024]
  float* buf1    = lds + 8 * H_;        // [8][1024]
  float* scratch = lds + 16 * H_;       // [1024]  wave0 neighbor staging

  const int wg   = blockIdx.x;          // 0..127
  const int w    = threadIdx.x >> 5;
  const int lane = threadIdx.x & 31;
  const int r    = wg * 8 + w;
  const int t    = r & (S_ - 1);

  // Per-lane parameter slices: features f = j*128 + 4*lane + c
  float4 w0v[8], w1v[8], bsv[8], lwv[8], lbv[8];
#pragma unroll
  for (int j = 0; j < 8; ++j) {
    int f = j * 128 + lane * 4;
    float4 a = *(const float4*)(Wsani + 2 * f);
    float4 b = *(const float4*)(Wsani + 2 * f + 4);
    w0v[j] = make_float4(a.x, a.z, b.x, b.z);
    w1v[j] = make_float4(a.y, a.w, b.y, b.w);
    bsv[j] = *(const float4*)(bsani + f);
    lwv[j] = *(const float4*)(lnw + f);
    lbv[j] = *(const float4*)(lnb + f);
  }

  // Load initial hidden row into buf0
#pragma unroll
  for (int j = 0; j < 8; ++j) {
    int f = j * 128 + lane * 4;
    *(float4*)(buf0 + w * H_ + f) = *(const float4*)(h0 + (size_t)r * H_ + f);
  }

  const int  tcons        = ((wg + 1) * 8) & (S_ - 1);  // t of consumer WG's first row
  const bool haveConsumer = (wg < 127) && (tcons != 0);

  // Publish version 0 (initial state of row 8*wg+7)
  if (w == 7 && haveConsumer) {
#pragma unroll
    for (int j = 0; j < 8; ++j) {
      int f = j * 128 + lane * 4;
      *(float4*)(ring + (size_t)wg * H_ + f) = *(float4*)(buf0 + w * H_ + f);
    }
    if (lane == 0)
      __hip_atomic_store(&ready[wg], 1, __ATOMIC_RELEASE, __HIP_MEMORY_SCOPE_AGENT);
  }
  __syncthreads();

  float4 val[8];
  for (int s = 0; s < S_; ++s) {
    float* cur = (s & 1) ? buf1 : buf0;
    float* nxt = (s & 1) ? buf0 : buf1;
    const bool need = (t > s);   // wave-uniform

    // Wave 0 stages the neighbor row (version s) from the global ring.
    if (w == 0 && need) {
      if (lane == 0) {
        while (__hip_atomic_load(&ready[wg - 1], __ATOMIC_ACQUIRE,
                                 __HIP_MEMORY_SCOPE_AGENT) < s + 1)
          __builtin_amdgcn_s_sleep(1);
      }
      __builtin_amdgcn_wave_barrier();
      const float* src = ring + (size_t)(s & 1) * 128 * H_ + (size_t)(wg - 1) * H_;
#if HAVE_ASYNC
#pragma unroll
      for (int j = 0; j < 8; ++j) {
        int f = j * 128 + lane * 4;
        __builtin_amdgcn_global_load_async_to_lds_b128(
            (v4i_as1*)(const_cast<float*>(src) + f),
            (v4i_as3*)(scratch + f), 0, 0);
      }
      __builtin_amdgcn_s_wait_asynccnt(0);
#else
#pragma unroll
      for (int j = 0; j < 8; ++j) {
        int f = j * 128 + lane * 4;
        *(float4*)(scratch + f) = *(const float4*)(src + f);
      }
#endif
      if (lane == 0)   // allow producer to overwrite this slot (after copy landed)
        __hip_atomic_store(&consumed[wg - 1], s + 1, __ATOMIC_RELEASE,
                           __HIP_MEMORY_SCOPE_AGENT);
    }

    const float* prevsrc = (w == 0) ? scratch : (cur + (w - 1) * H_);
    const float* own     = cur + w * H_;

    float sum = 0.f, sumsq = 0.f;
#pragma unroll
    for (int j = 0; j < 8; ++j) {
      int f = j * 128 + lane * 4;
      float4 hid = *(const float4*)(own + f);
      float4 o;
      if (need) {
        const float* src2 = (j < 4) ? prevsrc : own;          // cat = [prev | curr]
        int base = (j < 4) ? (2 * f) : (2 * f - H_);
        float4 a = *(const float4*)(src2 + base);
        float4 b = *(const float4*)(src2 + base + 4);
        o.x = fmaf(a.x, w0v[j].x, fmaf(a.y, w1v[j].x, bsv[j].x));
        o.y = fmaf(a.z, w0v[j].y, fmaf(a.w, w1v[j].y, bsv[j].y));
        o.z = fmaf(b.x, w0v[j].z, fmaf(b.y, w1v[j].z, bsv[j].z));
        o.w = fmaf(b.z, w0v[j].w, fmaf(b.w, w1v[j].w, bsv[j].w));
      } else {
        o = bsv[j];                                            // prev = curr = 0
      }
      float4 vv;
      vv.x = fmaf(fmaxf(o.x, 0.f), kSkip, hid.x);
      vv.y = fmaf(fmaxf(o.y, 0.f), kSkip, hid.y);
      vv.z = fmaf(fmaxf(o.z, 0.f), kSkip, hid.z);
      vv.w = fmaf(fmaxf(o.w, 0.f), kSkip, hid.w);
      val[j] = vv;
      sum   += vv.x + vv.y + vv.z + vv.w;
      sumsq += vv.x * vv.x + vv.y * vv.y + vv.z * vv.z + vv.w * vv.w;
    }

    // LayerNorm over H=1024: wave32 butterfly reduction (ds permute path)
#pragma unroll
    for (int off = 16; off >= 1; off >>= 1) {
      sum   += __shfl_xor(sum, off, 32);
      sumsq += __shfl_xor(sumsq, off, 32);
    }
    float mu   = sum * (1.0f / H_);
    float var  = sumsq * (1.0f / H_) - mu * mu;
    float rstd = rsqrtf(var + EPS_);
#pragma unroll
    for (int j = 0; j < 8; ++j) {
      int f = j * 128 + lane * 4;
      float4 nv;
      nv.x = fmaf((val[j].x - mu) * rstd, lwv[j].x, lbv[j].x);
      nv.y = fmaf((val[j].y - mu) * rstd, lwv[j].y, lbv[j].y);
      nv.z = fmaf((val[j].z - mu) * rstd, lwv[j].z, lbv[j].z);
      nv.w = fmaf((val[j].w - mu) * rstd, lwv[j].w, lbv[j].w);
      val[j] = nv;
      *(float4*)(nxt + w * H_ + f) = nv;
    }

    // Wave 7 publishes version s+1 for the consumer WG (with backpressure).
    int ver = s + 1;
    if (w == 7 && haveConsumer && ver < tcons) {
      if (lane == 0 && ver >= 2) {
        while (__hip_atomic_load(&consumed[wg], __ATOMIC_ACQUIRE,
                                 __HIP_MEMORY_SCOPE_AGENT) < ver - 1)
          __builtin_amdgcn_s_sleep(1);
      }
      __builtin_amdgcn_wave_barrier();
      float* dst = ring + (size_t)(ver & 1) * 128 * H_ + (size_t)wg * H_;
#pragma unroll
      for (int j = 0; j < 8; ++j) {
        int f = j * 128 + lane * 4;
        *(float4*)(dst + f) = val[j];
      }
      if (lane == 0)
        __hip_atomic_store(&ready[wg], ver + 1, __ATOMIC_RELEASE,
                           __HIP_MEMORY_SCOPE_AGENT);
    }
    __syncthreads();
  }

  // Final hidden state sits in buf0 -> emit pre-swizzled bf16 A operand for GEMM2
  for (int e = threadIdx.x; e < 8 * H_; e += 256) {
    int lr = e >> 10;
    int k  = e & 1023;
    int rr = wg * 8 + lr;
    hpack[packA_index(rr >> 4, k >> 5, rr & 15, k & 31)] = (__bf16)buf0[lr * H_ + k];
  }
}

// ---------------------------------------------------------------------------
// Loss: log_softmax over the S axis.  For each (b,e):
//   contribution = dot(x,y) - (sum_s x) * logsumexp_s(y)
// Online logsumexp, coalesced columns, deterministic 2-stage reduction.
// ---------------------------------------------------------------------------
__global__ void __launch_bounds__(256) loss_partial_kernel(const float* __restrict__ x,
                                                           const float* __restrict__ y,
                                                           float* __restrict__ partial) {
  int b = blockIdx.y;                            // 0..3
  int e = blockIdx.x * 256 + threadIdx.x;        // gridDim.x = 4 -> E columns
  float m = -3.402823466e38f, se = 0.f, dot = 0.f, sx = 0.f;
  for (int s = 0; s < S_; ++s) {
    size_t idx = ((size_t)(b * S_ + s)) * E_ + e;
    float yv = y[idx];
    float xv = x[idx];
    if (yv > m) { se = se * expf(m - yv) + 1.f; m = yv; }
    else        { se += expf(yv - m); }
    dot = fmaf(xv, yv, dot);
    sx += xv;
  }
  float lse = m + logf(se);
  float c = dot - sx * lse;

  __shared__ float red[256];
  red[threadIdx.x] = c;
  __syncthreads();
  for (int o = 128; o >= 1; o >>= 1) {
    if (threadIdx.x < o) red[threadIdx.x] += red[threadIdx.x + o];
    __syncthreads();
  }
  if (threadIdx.x == 0) partial[b * 4 + blockIdx.x] = red[0];
}

__global__ void loss_final_kernel(const float* __restrict__ partial, float* __restrict__ out) {
  if (threadIdx.x == 0) {
    float s = 0.f;
    for (int i = 0; i < 16; ++i) s += partial[i];
    out[0] = -s / (float)(B_ * E_);
  }
}

// ---------------------------------------------------------------------------
// Host orchestration (all on `stream`, graph-capture safe)
// ---------------------------------------------------------------------------
extern "C" void kernel_launch(void* const* d_in, const int* in_sizes, int n_in,
                              void* d_out, int out_size, void* d_ws, size_t ws_size,
                              hipStream_t stream) {
  const int*   labels = (const int*)d_in[0];
  const float* emb    = (const float*)d_in[1];
  const float* W_in   = (const float*)d_in[2];
  const float* b_in   = (const float*)d_in[3];
  const float* W_sani = (const float*)d_in[4];
  const float* b_sani = (const float*)d_in[5];
  const float* ln_w   = (const float*)d_in[6];
  const float* ln_b   = (const float*)d_in[7];
  const float* W_out  = (const float*)d_in[8];
  const float* b_out  = (const float*)d_in[9];
  float* out = (float*)d_out;

  char* ws = (char*)d_ws;
  const size_t MB = 1024 * 1024;
  float*  x_f32    = (float*) (ws + 0);          // 4 MB  [M, E]
  __bf16* xApack   = (__bf16*)(ws + 4  * MB);    // 2 MB
  __bf16* WinP     = (__bf16*)(ws + 6  * MB);    // 2 MB
  __bf16* WoutP    = (__bf16*)(ws + 8  * MB);    // 2 MB
  float*  h_f32    = (float*) (ws + 10 * MB);    // 4 MB  [M, H]
  __bf16* hApack   = (__bf16*)(ws + 14 * MB);    // 2 MB
  float*  y_f32    = (float*) (ws + 16 * MB);    // 4 MB  [M, E]
  float*  ring     = (float*) (ws + 20 * MB);    // 1 MB  [2][128][H]
  int*    ready    = (int*)   (ws + 21 * MB);
  int*    consumed = (int*)   (ws + 21 * MB + 4096);
  float*  partial  = (float*) (ws + 21 * MB + 8192);

  init_kernel<<<1, 256, 0, stream>>>(ready, consumed);
  pack_B_kernel<<<1024, 256, 0, stream>>>(W_in,  WinP);
  pack_B_kernel<<<1024, 256, 0, stream>>>(W_out, WoutP);
  gather_pack_kernel<<<1024, 256, 0, stream>>>(labels, emb, x_f32, xApack);

  wmma_gemm_bias_relu<<<dim3(8, 64), 256, 0, stream>>>(xApack, WinP, b_in, h_f32, H_);

  scan_kernel<<<128, 256, (17 * H_) * sizeof(float), stream>>>(
      h_f32, W_sani, b_sani, ln_w, ln_b, ring, ready, consumed, hApack);

  wmma_gemm_bias_relu<<<dim3(8, 64), 256, 0, stream>>>(hApack, WoutP, b_out, y_f32, E_);

  loss_partial_kernel<<<dim3(4, 4), 256, 0, stream>>>(x_f32, y_f32, partial);
  loss_final_kernel<<<1, 32, 0, stream>>>(partial, out);
}
